// HMM_30459908063824
// MI455X (gfx1250) — matrix operand order repaired and checked by reading, not compiled, not run on hardware
//
#include <hip/hip_runtime.h>
#include <hip/hip_bf16.h>
#include <math.h>

typedef _Float16 f16;
typedef __attribute__((ext_vector_type(16))) _Float16 v16h;
typedef __attribute__((ext_vector_type(8)))  _Float16 v8h;
typedef __attribute__((ext_vector_type(8)))  float    v8f;

#define B_    16
#define L_    1024
#define TL_   1026
#define D_    64
#define IN_   128
#define H_    512
#define NC_   32
#define OUTD_ 4096
#define M_    (B_*L_)            // 16384
#define LOG2PI_ 1.8378770664093453f

// ---------------- workspace layout (bytes) ----------------
static constexpr size_t OFF_XH  = 0;
static constexpr size_t SZ_XH   = (size_t)M_*IN_*2;       // 4 MiB f16
static constexpr size_t OFF_W0T = OFF_XH + SZ_XH;
static constexpr size_t SZ_W0T  = (size_t)H_*IN_*2;       // 128 KiB
static constexpr size_t OFF_W1T = OFF_W0T + SZ_W0T;
static constexpr size_t SZ_W1T  = (size_t)H_*H_*2;        // 512 KiB
static constexpr size_t OFF_W2T = OFF_W1T + SZ_W1T;
static constexpr size_t SZ_W2T  = (size_t)OUTD_*H_*2;     // 4 MiB
static constexpr size_t OFF_H1  = OFF_W2T + SZ_W2T;
static constexpr size_t SZ_H1   = (size_t)M_*H_*2;        // 16 MiB
static constexpr size_t OFF_H2  = OFF_H1 + SZ_H1;
static constexpr size_t SZ_H2   = SZ_H1;                  // 16 MiB
static constexpr size_t OFF_LP  = OFF_H2 + SZ_H2;
static constexpr size_t SZ_LP   = (size_t)M_*NC_*4;       // 2 MiB f32
static constexpr size_t OFF_PSI = OFF_LP + SZ_LP;         // B*L*NC int

// ---------------- CDNA5 async copy helpers ----------------
// GLOBAL_LOAD_ASYNC_TO_LDS_B128: LDS[vdst] = MEM[vaddr], tracked by ASYNCcnt
__device__ __forceinline__ void async_copy_b128(void* lds_dst, const void* gsrc) {
  unsigned lds = (unsigned)(size_t)lds_dst;  // flat shared addr[31:0] == LDS offset
  asm volatile("global_load_async_to_lds_b128 %0, %1, off"
               :: "v"(lds), "v"(gsrc) : "memory");
}
__device__ __forceinline__ void wait_async0() {
  asm volatile("s_wait_asynccnt 0x0" ::: "memory");
}

// ---------------- prep kernels ----------------
__global__ void build_xh(const float* __restrict__ x, f16* __restrict__ xh) {
  int idx = blockIdx.x*blockDim.x + threadIdx.x;
  if (idx >= M_*IN_) return;
  int row = idx >> 7, col = idx & 127;
  int b = row >> 10, t = row & 1023;
  float v = (col < 64) ? x[((size_t)(b*TL_ + t    ))*D_ + col]
                       : x[((size_t)(b*TL_ + t + 1))*D_ + (col - 64)];
  xh[idx] = (f16)v;
}

// W (K x N, f32, row-major) -> Wt (N x K, f16, row-major)
__global__ void transpose_cvt(const float* __restrict__ W, f16* __restrict__ Wt,
                              int K, int N) {
  int idx = blockIdx.x*blockDim.x + threadIdx.x;
  if (idx >= K*N) return;
  int n = idx / K, k = idx - n*K;
  Wt[idx] = (f16)W[(size_t)k*N + n];
}

// ---------------- WMMA fragment helpers ----------------
// A 16x32 f16: lane<16 row=lane, K = k..k+7 / k+16..k+23; lane>=16 shifted by 8
// B 32x16 f16: lane%16 = column; K = k..k+7 / k+16..k+23; lane>=16 shifted by 8
__device__ __forceinline__ v16h load_a_frag(const f16* Arow, int k, int hi) {
  v8h a0 = *(const v8h*)(Arow + k + hi*8);
  v8h a1 = *(const v8h*)(Arow + k + 16 + hi*8);
  v16h a;
#pragma unroll
  for (int i = 0; i < 8; ++i) { a[i] = a0[i]; a[8+i] = a1[i]; }
  return a;
}
__device__ __forceinline__ v16h load_b_frag(const f16* Bcol, int k, int hi) {
  v8h b0 = *(const v8h*)(Bcol + k + hi*8);
  v8h b1 = *(const v8h*)(Bcol + k + 16 + hi*8);
  v16h b;
#pragma unroll
  for (int i = 0; i < 8; ++i) { b[i] = b0[i]; b[8+i] = b1[i]; }
  return b;
}
__device__ __forceinline__ v8f wmma32(v16h a, v16h b, v8f c) {
  return __builtin_amdgcn_wmma_f32_16x16x32_f16(false, a, false, b,
                                                (short)0, c, false, false);
}

// ---------------- GEMM + bias + relu (f16 in/out, f32 acc) ----------------
// Block: 256 thr = 8 waves (2M x 4N), each wave a 2x2 grid of 16x16 tiles
// -> block tile 64(M) x 128(N). A block tile staged in LDS via async copies,
// double buffered; B fragments loaded straight from global (Wt is N x K).
template<int K, int N>
__global__ __launch_bounds__(256) void gemm_relu(
    const f16* __restrict__ A, const f16* __restrict__ Wt,
    const float* __restrict__ bias, f16* __restrict__ C) {
  __shared__ __align__(16) f16 A_lds[2][64][40];   // 40-half stride = 80B (pad)
  int tid  = threadIdx.x;
  int lane = tid & 31;
  int wave = tid >> 5;
  int wm = wave >> 2, wn = wave & 3;
  int blockRow = blockIdx.x*64;
  int rowBase  = blockRow + wm*32;
  int colBase  = blockIdx.y*128 + wn*32;
  int hi  = (lane >> 4) & 1;
  int l15 = lane & 15;

  // async stage: thread copies one 16B chunk: row cr, halves cc*8..cc*8+7
  int cr = tid >> 2, cc = tid & 3;
  const f16* Ag = A + (size_t)(blockRow + cr)*K + cc*8;

  const f16* Bcol0 = Wt + (size_t)(colBase + l15)*K;
  const f16* Bcol1 = Bcol0 + (size_t)16*K;

  v8f acc00 = {}, acc01 = {}, acc10 = {}, acc11 = {};

  async_copy_b128(&A_lds[0][cr][cc*8], Ag);
  constexpr int NS = K/32;
#pragma unroll
  for (int s = 0; s < NS; ++s) {
    wait_async0();
    __syncthreads();                       // buffer s ready for everyone
    if (s + 1 < NS)                        // overlap next copy with compute
      async_copy_b128(&A_lds[(s+1)&1][cr][cc*8], Ag + (s+1)*32);
    const int buf = s & 1;
    const int k   = s*32;
    v16h av0, av1;
    {
      v8h x0 = *(const v8h*)&A_lds[buf][wm*32 + l15     ][hi*8];
      v8h x1 = *(const v8h*)&A_lds[buf][wm*32 + l15     ][16 + hi*8];
      v8h y0 = *(const v8h*)&A_lds[buf][wm*32 + 16 + l15][hi*8];
      v8h y1 = *(const v8h*)&A_lds[buf][wm*32 + 16 + l15][16 + hi*8];
#pragma unroll
      for (int i = 0; i < 8; ++i) {
        av0[i] = x0[i]; av0[8+i] = x1[i];
        av1[i] = y0[i]; av1[8+i] = y1[i];
      }
    }
    v16h bv0 = load_b_frag(Bcol0, k, hi);
    v16h bv1 = load_b_frag(Bcol1, k, hi);
    acc00 = wmma32(av0, bv0, acc00);
    acc01 = wmma32(av0, bv1, acc01);
    acc10 = wmma32(av1, bv0, acc10);
    acc11 = wmma32(av1, bv1, acc11);
    __syncthreads();                       // reads done before buf is rewritten
  }
  // D layout: VGPR i -> row tileRow + i + hi*8, col tileCol + l15
#pragma unroll
  for (int i = 0; i < 8; ++i) {
    int r0 = rowBase + i + hi*8;
    int r1 = r0 + 16;
    int c0 = colBase + l15;
    int c1 = c0 + 16;
    float v;
    v = acc00[i] + bias[c0]; C[(size_t)r0*N + c0] = (f16)(v > 0.f ? v : 0.f);
    v = acc01[i] + bias[c1]; C[(size_t)r0*N + c1] = (f16)(v > 0.f ? v : 0.f);
    v = acc10[i] + bias[c0]; C[(size_t)r1*N + c0] = (f16)(v > 0.f ? v : 0.f);
    v = acc11[i] + bias[c1]; C[(size_t)r1*N + c1] = (f16)(v > 0.f ? v : 0.f);
  }
}

// ---------------- GEMM3 fused with Gaussian emission log-prob ----------------
// Block: 32 rows x 256 cols = TWO classes; 8 waves side-by-side in N, each
// wave a 2x2 grid of 16x16 tiles (4 WMMA per k-step). A tile (32x32) staged
// in LDS via async copies (double buffered, threads 0..127 copy one 16B chunk
// each); B fragments direct from W2t (N x K). Never materializes the 256 MB
// "out" tensor: fused Gaussian reduce writes lp (M x 32 f32).
__global__ __launch_bounds__(256) void gemm_emit(
    const f16* __restrict__ H2, const f16* __restrict__ W2t,
    const float* __restrict__ b2, const float* __restrict__ x,
    float* __restrict__ lp) {
  __shared__ __align__(16) f16 A_lds[2][32][40];
  __shared__ float outT[32][256];
  __shared__ float red[32][2][8];
  int tid  = threadIdx.x;
  int lane = tid & 31;
  int wn   = tid >> 5;                 // 0..7 : N strip
  int rowBase = blockIdx.x*32;
  int colBlk  = blockIdx.y*256;        // two classes: 2*blockIdx.y, +1
  int hi  = (lane >> 4) & 1;
  int l15 = lane & 15;

  // async stage of A tile: 32 rows x 32 halves = 128 chunks of 16B
  int cr = tid >> 2, cc = tid & 3;     // threads 0..127 active
  const f16* Ag = H2 + (size_t)(rowBase + cr)*H_ + cc*8;

  const f16* Bcol0 = W2t + (size_t)(colBlk + wn*32 + l15)*H_;
  const f16* Bcol1 = Bcol0 + (size_t)16*H_;

  v8f acc00 = {}, acc01 = {}, acc10 = {}, acc11 = {};

  if (tid < 128) async_copy_b128(&A_lds[0][cr][cc*8], Ag);
  constexpr int NS = H_/32;            // 16
#pragma unroll
  for (int s = 0; s < NS; ++s) {
    wait_async0();
    __syncthreads();
    if (s + 1 < NS && tid < 128)
      async_copy_b128(&A_lds[(s+1)&1][cr][cc*8], Ag + (s+1)*32);
    const int buf = s & 1;
    const int k   = s*32;
    v16h av0, av1;
    {
      v8h x0 = *(const v8h*)&A_lds[buf][l15     ][hi*8];
      v8h x1 = *(const v8h*)&A_lds[buf][l15     ][16 + hi*8];
      v8h y0 = *(const v8h*)&A_lds[buf][16 + l15][hi*8];
      v8h y1 = *(const v8h*)&A_lds[buf][16 + l15][16 + hi*8];
#pragma unroll
      for (int i = 0; i < 8; ++i) {
        av0[i] = x0[i]; av0[8+i] = x1[i];
        av1[i] = y0[i]; av1[8+i] = y1[i];
      }
    }
    v16h bv0 = load_b_frag(Bcol0, k, hi);
    v16h bv1 = load_b_frag(Bcol1, k, hi);
    acc00 = wmma32(av0, bv0, acc00);
    acc01 = wmma32(av0, bv1, acc01);
    acc10 = wmma32(av1, bv0, acc10);
    acc11 = wmma32(av1, bv1, acc11);
    __syncthreads();
  }
  // stage bias-added tile (32 x 256) in LDS
  int c0 = wn*32 + l15;
  int c1 = c0 + 16;
  float bb0 = b2[colBlk + c0];
  float bb1 = b2[colBlk + c1];
#pragma unroll
  for (int i = 0; i < 8; ++i) {
    outT[i + hi*8][c0]      = acc00[i] + bb0;
    outT[i + hi*8][c1]      = acc01[i] + bb1;
    outT[16 + i + hi*8][c0] = acc10[i] + bb0;
    outT[16 + i + hi*8][c1] = acc11[i] + bb1;
  }
  __syncthreads();
  // lp[row][cls] = -0.5 * sum_d((xt-mu)^2*exp(-lv) + lv + log2pi), 2 classes
  int row = tid >> 3;      // 0..31
  int seg = tid & 7;       // 8 dims each
  int grow = rowBase + row;
  int b = grow >> 10, t = grow & 1023;
  const float* xt = x + ((size_t)(b*TL_ + t + 2))*D_;
#pragma unroll
  for (int cl = 0; cl < 2; ++cl) {
    float p = 0.f;
#pragma unroll
    for (int q = 0; q < 8; ++q) {
      int d = seg*8 + q;
      float mu = outT[row][cl*128 + d];
      float lv = outT[row][cl*128 + 64 + d];
      float df = xt[d] - mu;
      p += df*df*__expf(-lv) + lv + LOG2PI_;
    }
    red[row][cl][seg] = p;
  }
  __syncthreads();
  if (seg == 0) {
#pragma unroll
    for (int cl = 0; cl < 2; ++cl) {
      float s = 0.f;
#pragma unroll
      for (int i = 0; i < 8; ++i) s += red[row][cl][i];
      lp[(size_t)grow*NC_ + blockIdx.y*2 + cl] = -0.5f*s;
    }
  }
}

// ---------------- forward + Viterbi scan (one wave32 per batch) ----------------
__global__ __launch_bounds__(32) void hmm_kernel(
    const float* __restrict__ lp, const float* __restrict__ logA,
    const float* __restrict__ logpi, float* __restrict__ out,
    int* __restrict__ psi) {
  __shared__ float sA[NC_*NC_];
  __shared__ float cprev[NC_];
  __shared__ float dprev[NC_];
  __shared__ float abuf[NC_];
  int j = threadIdx.x;   // state
  int b = blockIdx.x;    // batch
  {
    float m = -INFINITY;
    for (int i = 0; i < NC_; ++i) m = fmaxf(m, logA[j*NC_ + i]);
    float s = 0.f;
    for (int i = 0; i < NC_; ++i) s += __expf(logA[j*NC_ + i] - m);
    float lse = m + __logf(s);
    for (int i = 0; i < NC_; ++i) sA[j*NC_ + i] = logA[j*NC_ + i] - lse;
  }
  float mp = -INFINITY;
  for (int i = 0; i < NC_; ++i) mp = fmaxf(mp, logpi[i]);
  float sp = 0.f;
  for (int i = 0; i < NC_; ++i) sp += __expf(logpi[i] - mp);
  float pij = logpi[j] - (mp + __logf(sp));
  __syncthreads();

  const float* lpb = lp + (size_t)b*L_*NC_;
  float a0 = lpb[j] + pij;
  abuf[j] = a0;
  __syncthreads();
  float s0;
  {
    float m = -INFINITY;
    for (int i = 0; i < NC_; ++i) m = fmaxf(m, abuf[i]);
    float s = 0.f;
    for (int i = 0; i < NC_; ++i) s += __expf(abuf[i] - m);
    s0 = m + __logf(s);
  }
  float total = s0;
  __syncthreads();
  cprev[j] = a0 - s0;
  dprev[j] = a0;
  __syncthreads();

  for (int t = 1; t < L_; ++t) {
    float lpt = lpb[t*NC_ + j];
    float mm = -INFINITY;
    for (int i = 0; i < NC_; ++i) mm = fmaxf(mm, cprev[i] + sA[i*NC_ + j]);
    float ss = 0.f;
    for (int i = 0; i < NC_; ++i) ss += __expf(cprev[i] + sA[i*NC_ + j] - mm);
    float a = lpt + mm + __logf(ss);
    float best = -INFINITY; int arg = 0;
    for (int i = 0; i < NC_; ++i) {
      float v = dprev[i] + sA[i*NC_ + j];
      if (v > best) { best = v; arg = i; }
    }
    float dnew = best + lpt;
    psi[((size_t)(b*L_ + t))*NC_ + j] = arg;
    __syncthreads();
    abuf[j] = a;
    __syncthreads();
    float m2 = -INFINITY;
    for (int i = 0; i < NC_; ++i) m2 = fmaxf(m2, abuf[i]);
    float s2 = 0.f;
    for (int i = 0; i < NC_; ++i) s2 += __expf(abuf[i] - m2);
    float st = m2 + __logf(s2);
    total += st;
    __syncthreads();
    cprev[j] = a - st;
    dprev[j] = dnew;
    __syncthreads();
  }

  if (j == 0) {
    out[b] = total;                         // logp_x
    float best = -INFINITY; int cT = 0;
    for (int i = 0; i < NC_; ++i) if (dprev[i] > best) { best = dprev[i]; cT = i; }
    float* cout = out + B_;                 // c_est as floats
    cout[(size_t)b*L_ + (L_-1)] = (float)cT;
    int c = cT;
    for (int t = L_-1; t >= 1; --t) {
      c = psi[((size_t)(b*L_ + t))*NC_ + c];
      cout[(size_t)b*L_ + (t-1)] = (float)c;
    }
  }
}

extern "C" void kernel_launch(void* const* d_in, const int* in_sizes, int n_in,
                              void* d_out, int out_size, void* d_ws, size_t ws_size,
                              hipStream_t stream) {
  (void)in_sizes; (void)n_in; (void)out_size; (void)ws_size;
  const float* x     = (const float*)d_in[0];
  const float* w0    = (const float*)d_in[1];
  const float* b0    = (const float*)d_in[2];
  const float* w1    = (const float*)d_in[3];
  const float* b1    = (const float*)d_in[4];
  const float* w2    = (const float*)d_in[5];
  const float* b2    = (const float*)d_in[6];
  const float* logA  = (const float*)d_in[7];
  const float* logpi = (const float*)d_in[8];

  char* ws = (char*)d_ws;
  f16*   xh  = (f16*)(ws + OFF_XH);
  f16*   w0t = (f16*)(ws + OFF_W0T);
  f16*   w1t = (f16*)(ws + OFF_W1T);
  f16*   w2t = (f16*)(ws + OFF_W2T);
  f16*   h1  = (f16*)(ws + OFF_H1);
  f16*   h2  = (f16*)(ws + OFF_H2);
  float* lp  = (float*)(ws + OFF_LP);
  int*   psi = (int*)(ws + OFF_PSI);

  build_xh<<<(M_*IN_ + 255)/256, 256, 0, stream>>>(x, xh);
  transpose_cvt<<<(IN_*H_   + 255)/256, 256, 0, stream>>>(w0, w0t, IN_, H_);
  transpose_cvt<<<(H_*H_    + 255)/256, 256, 0, stream>>>(w1, w1t, H_, H_);
  transpose_cvt<<<(H_*OUTD_ + 255)/256, 256, 0, stream>>>(w2, w2t, H_, OUTD_);

  gemm_relu<IN_, H_><<<dim3(M_/64, H_/128), 256, 0, stream>>>(xh, w0t, b0, h1);
  gemm_relu<H_,  H_><<<dim3(M_/64, H_/128), 256, 0, stream>>>(h1, w1t, b1, h2);
  gemm_emit<<<dim3(M_/32, NC_/2), 256, 0, stream>>>(h2, w2t, b2, x, lp);
  hmm_kernel<<<B_, 32, 0, stream>>>(lp, logA, logpi, (float*)d_out, psi);
}
